// BiMambaEncoder_23003844837755
// MI455X (gfx1250) — matrix operand order repaired and checked by reading, not compiled, not run on hardware
//
#include <hip/hip_runtime.h>
#include <cstddef>
#include <cstdint>

// ---------------------------------------------------------------------------
// BiMamba encoder for gfx1250 (MI455X).  Dense projections use
// v_wmma_f32_16x16x32_bf16 (wave32 WMMA, f32 accumulate); conv/scan/LN are
// fp32 VALU.  B=8, T=1024, D_MODEL=256, DI=512, D_STATE=16, DT_RANK=16, NB=4.
// ---------------------------------------------------------------------------

typedef __bf16 bf16_t;
typedef __attribute__((ext_vector_type(16))) __bf16 v16bf;
typedef __attribute__((ext_vector_type(8)))  float  v8f;

#define BATCH   8
#define SEQ     1024
#define DMODEL  256
#define DI      512
#define DSTATE  16
#define DTRANK  16
#define NLAYER  4
#define MROWS   (BATCH * SEQ)           // 8192
#define XDBLW   (DTRANK + 2 * DSTATE)   // 48

enum { EPI_NONE = 0, EPI_BIAS = 1, EPI_BIAS_SOFTPLUS = 2 };

// ---------------------------------------------------------------------------
// bf16 GEMM:  C[M,N] = epi( A[M,K] @ W[N,K]^T + bias )
// One wave computes a 16x(NT*16) strip; NT and the epilogue are template
// parameters so every accumulator index is a compile-time constant (fixed,
// even-aligned VGPR octets -> clean back-to-back v_wmma, no movrel traffic).
// Fragment k order: contiguous k = 16*half + i in A and B (wave32, 16-bit A 16x32 / B 32x16).
// ---------------------------------------------------------------------------
template<int NT, int EPI>
__global__ __launch_bounds__(128)
void bm_gemm_bf16_wmma(const bf16_t* __restrict__ A,
                       const bf16_t* __restrict__ W,
                       const float*  __restrict__ bias,
                       float* __restrict__ C,
                       int N, int K)
{
    const int lane = threadIdx.x & 31;
    const int wave = threadIdx.x >> 5;
    const int mt   = blockIdx.x * 4 + wave;     // 16-row tile index (M=8192 exact)
    const int nt0  = blockIdx.y * NT;           // first 16-col tile index
    const int half = lane >> 4;                 // 0 | 1
    const int mr   = lane & 15;

    v8f acc[NT];
#pragma unroll
    for (int j = 0; j < NT; ++j)
#pragma unroll
        for (int r = 0; r < 8; ++r) acc[j][r] = 0.0f;

    const bf16_t* arow = A + (size_t)(mt * 16 + mr) * K;

    for (int k0 = 0; k0 < K; k0 += 32) {
        // --- A fragment: contiguous K run of 16 per lane half (k = 16*half + i) ---
        v16bf afrag;
#pragma unroll
        for (int i = 0; i < 8; ++i) afrag[i]     = arow[k0 + 16 * half + i];
#pragma unroll
        for (int i = 0; i < 8; ++i) afrag[i + 8] = arow[k0 + 16 * half + 8 + i];

#pragma unroll
        for (int j = 0; j < NT; ++j) {
            // --- B fragment: column n of W^T == contiguous row n of W ---
            const bf16_t* wrow =
                W + (size_t)((nt0 + j) * 16 + mr) * K + k0 + 16 * half;
            v16bf bfrag;
#pragma unroll
            for (int i = 0; i < 16; ++i) bfrag[i] = wrow[i];

            acc[j] = __builtin_amdgcn_wmma_f32_16x16x32_bf16(
                         false, afrag, false, bfrag, (short)0, acc[j],
                         false, false);
        }
    }

#pragma unroll
    for (int j = 0; j < NT; ++j) {
        const int col = (nt0 + j) * 16 + mr;
        float bv = 0.0f;
        if (EPI != EPI_NONE) bv = bias[col];
#pragma unroll
        for (int r = 0; r < 8; ++r) {
            const int row = mt * 16 + half * 8 + r;
            float v = acc[j][r] + bv;
            if (EPI == EPI_BIAS_SOFTPLUS)
                v = (v > 20.0f) ? v : log1pf(__expf(v));
            C[(size_t)row * N + col] = v;
        }
    }
}

template<int NT, int EPI>
static inline void launch_gemm(const bf16_t* A, const bf16_t* W,
                               const float* bias, float* C,
                               int N, int K, hipStream_t stream)
{
    dim3 grid(MROWS / 64, N / (NT * 16));
    bm_gemm_bf16_wmma<NT, EPI><<<grid, 128, 0, stream>>>(A, W, bias, C, N, K);
}

// ---------------------------------------------------------------------------
// Depthwise causal conv (4 taps) + SiLU.  Writes fp32 + bf16 copies.
// ---------------------------------------------------------------------------
__global__ void bm_conv_silu(const float* __restrict__ xz,
                             const float* __restrict__ cw,
                             const float* __restrict__ cb,
                             float* __restrict__ xc,
                             bf16_t* __restrict__ xc_bf,
                             int layer)
{
    const int i = blockIdx.x * blockDim.x + threadIdx.x;
    if (i >= MROWS * DI) return;
    const int d   = i & (DI - 1);
    const int row = i >> 9;
    const int b   = row >> 10;
    const int t   = row & (SEQ - 1);
    const float* w = cw + ((size_t)layer * DI + d) * 4;
    float acc = cb[layer * DI + d];
#pragma unroll
    for (int k = 0; k < 4; ++k) {
        const int tt = t - 3 + k;
        if (tt >= 0)
            acc += w[k] * xz[((size_t)b * SEQ + tt) * (2 * DI) + d];
    }
    const float v = acc / (1.0f + __expf(-acc));   // silu
    xc[i]    = v;
    xc_bf[i] = (bf16_t)v;
}

// ---------------------------------------------------------------------------
// Pack dt input: first 16 cols of xdbl, zero-padded to K=32 (bf16).
// ---------------------------------------------------------------------------
__global__ void bm_pack_dtin(const float* __restrict__ xdbl,
                             bf16_t* __restrict__ dst)
{
    const int i = blockIdx.x * blockDim.x + threadIdx.x;
    if (i >= MROWS * 32) return;
    const int k = i & 31;
    const int r = i >> 5;
    dst[i] = (k < DTRANK) ? (bf16_t)xdbl[(size_t)r * XDBLW + k] : (bf16_t)0.0f;
}

// Pad dt_proj weights (NB,512,16) -> (NB,512,32) bf16.
__global__ void bm_pack_dtw(const float* __restrict__ w, bf16_t* __restrict__ dst)
{
    const int i = blockIdx.x * blockDim.x + threadIdx.x;
    if (i >= NLAYER * DI * 32) return;
    const int k = i & 31;
    const int r = i >> 5;
    dst[i] = (k < DTRANK) ? (bf16_t)w[(size_t)r * DTRANK + k] : (bf16_t)0.0f;
}

__global__ void bm_f32_to_bf16(const float* __restrict__ s,
                               bf16_t* __restrict__ d, int n)
{
    const int i = blockIdx.x * blockDim.x + threadIdx.x;
    if (i < n) d[i] = (bf16_t)s[i];
}

// copy fp32 + write bf16 shadow (used to seed the residual stream with x)
__global__ void bm_copy_f32_bf(const float* __restrict__ s,
                               float* __restrict__ d,
                               bf16_t* __restrict__ dbf, int n)
{
    const int i = blockIdx.x * blockDim.x + threadIdx.x;
    if (i < n) { const float v = s[i]; d[i] = v; dbf[i] = (bf16_t)v; }
}

// reverse along T, writing fp32 + bf16 shadow
__global__ void bm_reverse_f32_bf(const float* __restrict__ s,
                                  float* __restrict__ d,
                                  bf16_t* __restrict__ dbf)
{
    const int i = blockIdx.x * blockDim.x + threadIdx.x;
    if (i >= MROWS * DMODEL) return;
    const int c   = i & (DMODEL - 1);
    const int row = i >> 8;
    const int b   = row >> 10;
    const int t   = row & (SEQ - 1);
    const float v = s[((size_t)b * SEQ + (SEQ - 1 - t)) * DMODEL + c];
    d[i] = v; dbf[i] = (bf16_t)v;
}

// forward result -> left half of concat buffer (bf16)
__global__ void bm_copy_left(const bf16_t* __restrict__ s,
                             bf16_t* __restrict__ comb)
{
    const int i = blockIdx.x * blockDim.x + threadIdx.x;
    if (i >= MROWS * DMODEL) return;
    const int c = i & (DMODEL - 1);
    const int r = i >> 8;
    comb[(size_t)r * (2 * DMODEL) + c] = s[i];
}

// backward result (fp32, still reversed) -> re-reverse into right half (bf16)
__global__ void bm_reverse_right(const float* __restrict__ s,
                                 bf16_t* __restrict__ comb)
{
    const int i = blockIdx.x * blockDim.x + threadIdx.x;
    if (i >= MROWS * DMODEL) return;
    const int c   = i & (DMODEL - 1);
    const int row = i >> 8;
    const int b   = row >> 10;
    const int t   = row & (SEQ - 1);
    const float v = s[((size_t)b * SEQ + (SEQ - 1 - t)) * DMODEL + c];
    comb[(size_t)row * (2 * DMODEL) + DMODEL + c] = (bf16_t)v;
}

// ---------------------------------------------------------------------------
// Selective scan: one thread per (b,d) channel, 16 states in registers,
// 1024 serial steps.  Fuses +xc*D and *silu(z); emits bf16 for the out-proj.
// ---------------------------------------------------------------------------
__global__ __launch_bounds__(256)
void bm_scan(const float* __restrict__ dt,
             const float* __restrict__ xc,
             const float* __restrict__ xz,
             const float* __restrict__ xdbl,
             const float* __restrict__ alog,
             const float* __restrict__ dsk,
             bf16_t* __restrict__ ybf)
{
    const int idx = blockIdx.x * 256 + threadIdx.x;   // 0..4095
    if (idx >= BATCH * DI) return;
    const int d = idx & (DI - 1);
    const int b = idx >> 9;

    float Ac[DSTATE];
#pragma unroll
    for (int n = 0; n < DSTATE; ++n) Ac[n] = -__expf(alog[d * DSTATE + n]);
    const float Dd = dsk[d];

    float h[DSTATE];
#pragma unroll
    for (int n = 0; n < DSTATE; ++n) h[n] = 0.0f;

    for (int t = 0; t < SEQ; ++t) {
        const size_t row = (size_t)b * SEQ + t;
        const float dtv = dt[row * DI + d];
        const float xcv = xc[row * DI + d];
        const float zv  = xz[row * (2 * DI) + DI + d];
        const float* bsc = xdbl + row * XDBLW + DTRANK;  // Bs[16] then Cs[16]
        const float dbx = dtv * xcv;
        float y = 0.0f;
#pragma unroll
        for (int n = 0; n < DSTATE; ++n) {
            const float dA = __expf(dtv * Ac[n]);
            h[n] = dA * h[n] + dbx * bsc[n];
            y += h[n] * bsc[DSTATE + n];
        }
        y += xcv * Dd;
        y *= zv / (1.0f + __expf(-zv));                  // * silu(z)
        ybf[row * DI + d] = (bf16_t)y;
    }
}

// ---------------------------------------------------------------------------
// NaN flag reduction over the projected mamba output.
// ---------------------------------------------------------------------------
__global__ void bm_nan_or(const float* __restrict__ v, int n,
                          int* __restrict__ flag)
{
    const int i = blockIdx.x * blockDim.x + threadIdx.x;
    if (i < n) { const float x = v[i]; if (x != x) atomicOr(flag, 1); }
}

// ---------------------------------------------------------------------------
// Residual + LayerNorm (gated by NaN flag).  One 256-thread block per row.
// ---------------------------------------------------------------------------
__global__ __launch_bounds__(256)
void bm_ln_residual(const float* __restrict__ yproj,
                    const float* __restrict__ g,
                    const float* __restrict__ bvec,
                    const int*  __restrict__ nanflag,
                    float* __restrict__ cur,
                    bf16_t* __restrict__ cur_bf)
{
    __shared__ float s1[256];
    __shared__ float s2[256];
    const int row = blockIdx.x;
    const int c   = threadIdx.x;
    const size_t idx = (size_t)row * DMODEL + c;
    const float hold = cur[idx];
    const float x = yproj[idx] + hold;
    s1[c] = x; s2[c] = x * x;
    __syncthreads();
    for (int off = 128; off > 0; off >>= 1) {
        if (c < off) { s1[c] += s1[c + off]; s2[c] += s2[c + off]; }
        __syncthreads();
    }
    const float mu  = s1[0] * (1.0f / DMODEL);
    const float var = s2[0] * (1.0f / DMODEL) - mu * mu;
    const float out = (x - mu) * rsqrtf(var + 1e-5f) * g[c] + bvec[c];
    const float res = (*nanflag) ? hold : out;
    cur[idx]    = res;
    cur_bf[idx] = (bf16_t)res;
}

// ---------------------------------------------------------------------------
// Host orchestration
// ---------------------------------------------------------------------------
static inline size_t bm_align(size_t x) { return (x + 255) & ~(size_t)255; }

extern "C" void kernel_launch(void* const* d_in, const int* in_sizes, int n_in,
                              void* d_out, int out_size, void* d_ws, size_t ws_size,
                              hipStream_t stream)
{
    (void)in_sizes; (void)n_in; (void)out_size; (void)ws_size;
    const float* x      = (const float*)d_in[0];
    const float* in_w   = (const float*)d_in[1];
    const float* conv_w = (const float*)d_in[2];
    const float* conv_b = (const float*)d_in[3];
    const float* xp_w   = (const float*)d_in[4];
    const float* dt_w   = (const float*)d_in[5];
    const float* dt_b   = (const float*)d_in[6];
    const float* a_log  = (const float*)d_in[7];
    const float* dsk    = (const float*)d_in[8];
    const float* out_w  = (const float*)d_in[9];
    const float* ln_g   = (const float*)d_in[10];
    const float* ln_b   = (const float*)d_in[11];
    const float* fuse_w = (const float*)d_in[12];
    const float* fuse_b = (const float*)d_in[13];
    float* out = (float*)d_out;

    // ---- workspace layout (bump allocator) ----
    uint8_t* p = (uint8_t*)d_ws;
    size_t off = 0;
    auto alloc = [&](size_t bytes) { void* r = p + off; off = bm_align(off + bytes); return r; };
    float*  xz      = (float*)  alloc((size_t)MROWS * 2 * DI * 4);
    float*  xc      = (float*)  alloc((size_t)MROWS * DI * 4);
    bf16_t* xc_bf   = (bf16_t*) alloc((size_t)MROWS * DI * 2);
    float*  xdbl    = (float*)  alloc((size_t)MROWS * XDBLW * 4);
    bf16_t* dtin_bf = (bf16_t*) alloc((size_t)MROWS * 32 * 2);
    float*  dt      = (float*)  alloc((size_t)MROWS * DI * 4);
    bf16_t* y_bf    = (bf16_t*) alloc((size_t)MROWS * DI * 2);
    float*  yproj   = (float*)  alloc((size_t)MROWS * DMODEL * 4);
    float*  cur     = (float*)  alloc((size_t)MROWS * DMODEL * 4);
    bf16_t* cur_bf  = (bf16_t*) alloc((size_t)MROWS * DMODEL * 2);
    bf16_t* comb_bf = (bf16_t*) alloc((size_t)MROWS * 2 * DMODEL * 2);
    bf16_t* inw_bf  = (bf16_t*) alloc((size_t)NLAYER * 2 * DI * DMODEL * 2);
    bf16_t* xpw_bf  = (bf16_t*) alloc((size_t)NLAYER * XDBLW * DI * 2);
    bf16_t* dtw_bf  = (bf16_t*) alloc((size_t)NLAYER * DI * 32 * 2);
    bf16_t* ow_bf   = (bf16_t*) alloc((size_t)NLAYER * DMODEL * DI * 2);
    bf16_t* fw_bf   = (bf16_t*) alloc((size_t)DMODEL * 2 * DMODEL * 2);
    int*    flag    = (int*)    alloc(256);

    const int TPB = 256;
    auto blks = [&](int n) { return (n + TPB - 1) / TPB; };

    // ---- one-time (per launch) weight conversion ----
    { int n = NLAYER * 2 * DI * DMODEL;
      bm_f32_to_bf16<<<blks(n), TPB, 0, stream>>>(in_w, inw_bf, n); }
    { int n = NLAYER * XDBLW * DI;
      bm_f32_to_bf16<<<blks(n), TPB, 0, stream>>>(xp_w, xpw_bf, n); }
    { int n = NLAYER * DI * 32;
      bm_pack_dtw<<<blks(n), TPB, 0, stream>>>(dt_w, dtw_bf); }
    { int n = NLAYER * DMODEL * DI;
      bm_f32_to_bf16<<<blks(n), TPB, 0, stream>>>(out_w, ow_bf, n); }
    { int n = DMODEL * 2 * DMODEL;
      bm_f32_to_bf16<<<blks(n), TPB, 0, stream>>>(fuse_w, fw_bf, n); }

    auto run_layer = [&](int j) {
        // 1) in_proj: xz = cur @ in_w[j]^T   (K=256, N=1024, 4 tiles/wave)
        launch_gemm<4, EPI_NONE>(cur_bf, inw_bf + (size_t)j * 2 * DI * DMODEL,
                                 nullptr, xz, 2 * DI, DMODEL, stream);
        // 2) depthwise causal conv + silu
        bm_conv_silu<<<blks(MROWS * DI), TPB, 0, stream>>>(
            xz, conv_w, conv_b, xc, xc_bf, j);
        // 3) x_proj: xdbl = xc @ xp_w[j]^T   (K=512, N=48, 3 tiles/wave)
        launch_gemm<3, EPI_NONE>(xc_bf, xpw_bf + (size_t)j * XDBLW * DI,
                                 nullptr, xdbl, XDBLW, DI, stream);
        // 4) pack dt input (16 -> pad 32)
        bm_pack_dtin<<<blks(MROWS * 32), TPB, 0, stream>>>(xdbl, dtin_bf);
        // 5) dt = softplus(dtin @ dt_w[j]^T + dt_b[j])   (K=32, N=512)
        launch_gemm<4, EPI_BIAS_SOFTPLUS>(dtin_bf, dtw_bf + (size_t)j * DI * 32,
                                          dt_b + (size_t)j * DI, dt,
                                          DI, 32, stream);
        // 6) selective scan (+ D skip, * silu(z))
        bm_scan<<<(BATCH * DI) / 256, 256, 0, stream>>>(
            dt, xc, xz, xdbl, a_log + (size_t)j * DI * DSTATE,
            dsk + (size_t)j * DI, y_bf);
        // 7) out_proj: yproj = y @ out_w[j]^T   (K=512, N=256)
        launch_gemm<4, EPI_NONE>(y_bf, ow_bf + (size_t)j * DMODEL * DI,
                                 nullptr, yproj, DMODEL, DI, stream);
        // 8) NaN guard flag
        hipMemsetAsync(flag, 0, sizeof(int), stream);
        bm_nan_or<<<blks(MROWS * DMODEL), TPB, 0, stream>>>(
            yproj, MROWS * DMODEL, flag);
        // 9) residual + LayerNorm
        bm_ln_residual<<<MROWS, DMODEL, 0, stream>>>(
            yproj, ln_g + (size_t)j * DMODEL, ln_b + (size_t)j * DMODEL,
            flag, cur, cur_bf);
    };

    const int NX = MROWS * DMODEL;

    // ---- forward stack ----
    bm_copy_f32_bf<<<blks(NX), TPB, 0, stream>>>(x, cur, cur_bf, NX);
    run_layer(0);
    run_layer(1);
    bm_copy_left<<<blks(NX), TPB, 0, stream>>>(cur_bf, comb_bf);

    // ---- backward stack (physically reversed sequence) ----
    bm_reverse_f32_bf<<<blks(NX), TPB, 0, stream>>>(x, cur, cur_bf);
    run_layer(2);
    run_layer(3);
    bm_reverse_right<<<blks(NX), TPB, 0, stream>>>(cur, comb_bf);

    // ---- fuse: out = comb @ fuse_w^T + fuse_b   (K=512, N=256) ----
    launch_gemm<4, EPI_BIAS>(comb_bf, fw_bf, fuse_b, out,
                             DMODEL, 2 * DMODEL, stream);
}
